// CapHLA_EL_25915832664128
// MI455X (gfx1250) — matrix operand order repaired and checked by reading, not compiled, not run on hardware
//
#include <hip/hip_runtime.h>

// ---------------- model constants ----------------
#define BSZ   1024
#define SEQ   59
#define VOC   21
#define CCH   3200      // C
#define KW    9
#define NHEAD 9
#define HDIM  189       // V*H
#define FLATN 1239      // 21*59
#define FLATP 1248      // padded to mult of 32
#define NHID  800
#define EPSC  1e-5f

typedef __attribute__((ext_vector_type(16))) __bf16 v16b;
typedef __attribute__((ext_vector_type(8)))  __bf16 v8b;
typedef __attribute__((ext_vector_type(8)))  float  v8f;
typedef __attribute__((ext_vector_type(4)))  unsigned int v4u;
typedef __attribute__((ext_vector_type(8)))  int v8i;
typedef __attribute__((ext_vector_type(4)))  int v4i;

#if defined(__has_builtin)
#if __has_builtin(__builtin_amdgcn_tensor_load_to_lds) && __has_builtin(__builtin_amdgcn_s_wait_tensorcnt)
#define HAVE_TDM 1
#endif
#endif
#ifndef HAVE_TDM
#define HAVE_TDM 0
#endif

__device__ __forceinline__ float sigf(float x) { return 1.f / (1.f + __expf(-x)); }
__device__ __forceinline__ float siluf(float x) { return x * sigf(x); }

// ---- WMMA bf16 16x16x32, fp32 accum ----
__device__ __forceinline__ v8f wmma_bf(v16b a, v16b b, v8f c) {
    return __builtin_amdgcn_wmma_f32_16x16x32_bf16(false, a, false, b, (short)0, c, false, false);
}

// A fragment: 16x32 tile, row-major base[row*lda + k], lda mult of 8, ko mult of 32.
__device__ __forceinline__ v16b frag_A(const __bf16* base, int lda, int ko) {
    int lane = threadIdx.x & 31;
    int row = lane & 15, half = lane >> 4;
    const __bf16* p = base + row * lda + ko + half * 8;
    v8b lo = *(const v8b*)(p);
    v8b hi = *(const v8b*)(p + 16);
    return __builtin_shufflevector(lo, hi, 0,1,2,3,4,5,6,7,8,9,10,11,12,13,14,15);
}

// B fragment: B^T stored [n*ldb + k]; lane n = lane&15, k = 8*half+s / 16+8*half+(s-8).
__device__ __forceinline__ v16b frag_B(const __bf16* base, int ldb, int ko) {
    int lane = threadIdx.x & 31;
    int n = lane & 15;
    const __bf16* p = base + n * ldb + ko + ((lane >> 4) << 3);
    v8b lo = *(const v8b*)(p);
    v8b hi = *(const v8b*)(p + 16);
    return __builtin_shufflevector(lo, hi, 0,1,2,3,4,5,6,7,8,9,10,11,12,13,14,15);
}

#if HAVE_TDM
// Tensor Data Mover: 2D bf16 tile (tile_x contiguous elems per row, tile_y rows,
// row pitch = stride elems) from global -> LDS (packed rows). D# per ISA ch.8.
__device__ __forceinline__ void tdm_load_2d(__bf16* lds_dst, const __bf16* gsrc,
                                            unsigned tile_x, unsigned tile_y,
                                            unsigned stride, unsigned td0, unsigned td1) {
    unsigned long long ga = (unsigned long long)(size_t)(const void*)gsrc;
    unsigned ga_lo = (unsigned)__builtin_amdgcn_readfirstlane((int)(unsigned)ga);
    unsigned ga_hi = (unsigned)__builtin_amdgcn_readfirstlane((int)(unsigned)(ga >> 32));
    unsigned lds   = (unsigned)__builtin_amdgcn_readfirstlane(
                         (int)(unsigned)(unsigned long long)(size_t)(void*)lds_dst);
    // group0: [1:0]=count=1 | [63:32]=lds_addr | [120:64]=global_addr | [127:126]=type 2
    v4u g0 = { 1u, lds, ga_lo, (ga_hi & 0x1FFFFFFu) | (2u << 30) };
    // group1: data_size=1(2B) @ [17:16]; tensor_dim0 @ [79:48]; tensor_dim1 @ [111:80];
    // tile_dim0 @ [127:112]; tile_dim1 @ [143:128]; tensor_dim0_stride @ [207:160]
    v8i g1 = { (int)(1u << 16),
               (int)((td0 & 0xFFFFu) << 16),
               (int)(((td0 >> 16) & 0xFFFFu) | ((td1 & 0xFFFFu) << 16)),
               (int)(((td1 >> 16) & 0xFFFFu) | ((tile_x & 0xFFFFu) << 16)),
               (int)(tile_y & 0xFFFFu),
               (int)stride, 0, 0 };
    v4i z4 = { 0, 0, 0, 0 };
#if __clang_major__ >= 23
    v8i z8 = { 0, 0, 0, 0, 0, 0, 0, 0 };
    __builtin_amdgcn_tensor_load_to_lds(g0, g1, z4, z4, z8, 0);
#else
    __builtin_amdgcn_tensor_load_to_lds(g0, g1, z4, z4, 0);
#endif
}
__device__ __forceinline__ void tdm_load_1d(__bf16* lds_dst, const __bf16* gsrc, unsigned n) {
    tdm_load_2d(lds_dst, gsrc, n, 1, n, n, 1);
}
__device__ __forceinline__ void tdm_wait() {
    __builtin_amdgcn_s_wait_tensorcnt(0);
    __builtin_amdgcn_wave_barrier();
}
#endif

// =======================================================================
// prep: convert/transposed/padded bf16 weight tensors
//  W1p  [6400][32]  (k padded 21->32)
//  W2t  [32][3200]  (rows = out vocab padded, cols = channel)
//  Wqkvt[3][9][32][32] (B^T per head: n=head dim, k=vocab, both padded)
//  Wotp [32][192]   (B^T of Wo)
//  Wf1t [800][1248], Wf2t [64][800]
// =======================================================================
#define PA1 (6400 * 32)
#define PA2 (PA1 + 32 * CCH)
#define PA3 (PA2 + 3 * NHEAD * 1024)
#define PA4 (PA3 + 32 * 192)
#define PA5 (PA4 + NHID * FLATP)
#define PA6 (PA5 + 64 * NHID)

__global__ __launch_bounds__(256) void k_prep(
    const float* __restrict__ W1, const float* __restrict__ W2,
    const float* __restrict__ Wq, const float* __restrict__ Wk,
    const float* __restrict__ Wv, const float* __restrict__ Wo,
    const float* __restrict__ Wf1, const float* __restrict__ Wf2,
    __bf16* __restrict__ W1p, __bf16* __restrict__ W2t,
    __bf16* __restrict__ Wqkvt, __bf16* __restrict__ Wotp,
    __bf16* __restrict__ Wf1t, __bf16* __restrict__ Wf2t) {
    int i = blockIdx.x * 256 + threadIdx.x;
    if (i < PA1) {
        int o = i >> 5, k = i & 31;
        W1p[i] = (k < VOC) ? (__bf16)W1[o * VOC + k] : (__bf16)0.f;
    } else if (i < PA2) {
        int j = i - PA1; int n = j / CCH, c = j % CCH;
        W2t[j] = (n < VOC) ? (__bf16)W2[n * CCH + c] : (__bf16)0.f;
    } else if (i < PA3) {
        int j = i - PA2; int m = j / (NHEAD * 1024); int r = j % (NHEAD * 1024);
        int h = r >> 10, q = r & 1023, n = q >> 5, k = q & 31;
        const float* Wm = (m == 0) ? Wq : ((m == 1) ? Wk : Wv);
        Wqkvt[j] = (n < VOC && k < VOC) ? (__bf16)Wm[k * HDIM + h * VOC + n] : (__bf16)0.f;
    } else if (i < PA4) {
        int j = i - PA3; int n = j / 192, k = j % 192;
        Wotp[j] = (n < VOC && k < HDIM) ? (__bf16)Wo[k * VOC + n] : (__bf16)0.f;
    } else if (i < PA5) {
        int j = i - PA4; int n = j / FLATP, k = j % FLATP;
        Wf1t[j] = (k < FLATN) ? (__bf16)Wf1[k * NHID + n] : (__bf16)0.f;
    } else if (i < PA6) {
        int j = i - PA5; int n = j / NHID, k = j % NHID;
        Wf2t[j] = (__bf16)Wf2[k * 64 + n];
    }
}

// =======================================================================
// conv module (GLU conv block + residual + LN), one WG per batch element.
// 256 threads = 8 waves; 32-channel chunks per wave, weights via TDM.
// =======================================================================
__global__ __launch_bounds__(256) void k_conv(
    const float* __restrict__ pep, const float* __restrict__ mhc,
    const __bf16* __restrict__ W1p, const float* __restrict__ b1,
    const float* __restrict__ Wd,  const float* __restrict__ bd,
    const float* __restrict__ g1,  const float* __restrict__ be1,
    const float* __restrict__ m1,  const float* __restrict__ v1,
    const __bf16* __restrict__ W2t, const float* __restrict__ b2,
    const float* __restrict__ lng, const float* __restrict__ lnb,
    float* __restrict__ X1f, __bf16* __restrict__ X1a) {
    __shared__ __align__(16) float  Xf[64 * 32];
    __shared__ __align__(16) __bf16 Xa[64 * 32];
    __shared__ __align__(16) float  yacc[64 * 32];
    __shared__ __align__(16) __bf16 Pw1a[8][32 * 32];
    __shared__ __align__(16) __bf16 Pw1g[8][32 * 32];
    __shared__ __align__(16) __bf16 Phg[8][64 * 32];
    __shared__ __align__(16) __bf16 Psa[8][64 * 32];
    __shared__ __align__(16) __bf16 Pw2[8][32 * 32];

    int b = blockIdx.x, tid = threadIdx.x, w = tid >> 5, lane = tid & 31;
    int nl = lane & 15, hi8 = (lane >> 4) << 3;

    for (int i = tid; i < 64 * 32; i += 256) {
        int r = i >> 5, c = i & 31;
        float v = 0.f;
        if (c < VOC && r < SEQ)
            v = (r < 15) ? pep[(b * 15 + r) * VOC + c] : mhc[(b * 44 + (r - 15)) * VOC + c];
        Xf[i] = v; Xa[i] = (__bf16)v;
        yacc[i] = (c < VOC) ? b2[c] : 0.f;
    }
    __syncthreads();

    v16b Af[4];
#pragma unroll
    for (int mt = 0; mt < 4; ++mt) Af[mt] = frag_A(Xa + mt * 16 * 32, 32, 0);

    v8f yfr[4][2] = {};

    for (int it = 0; it < 13; ++it) {
        int cb = (w + 8 * it) * 32;           // wave-uniform channel chunk base
        if (cb < CCH) {
            // stage bf16 weight tiles into this wave's LDS region
#if HAVE_TDM
            tdm_load_2d(Pw1a[w], W1p + (size_t)cb * 32,          32, 32, 32, 32, 32);
            tdm_load_2d(Pw1g[w], W1p + (size_t)(CCH + cb) * 32,  32, 32, 32, 32, 32);
            tdm_load_2d(Pw2[w],  W2t + cb,                       32, 32, CCH, 32, 32);
            tdm_wait();
#else
            for (int i = lane; i < 32 * 32; i += 32) {
                int n = i >> 5, k = i & 31;
                Pw1a[w][i] = W1p[(size_t)(cb + n) * 32 + k];
                Pw1g[w][i] = W1p[(size_t)(CCH + cb + n) * 32 + k];
                Pw2[w][i]  = W2t[(size_t)n * CCH + cb + k];
            }
            __builtin_amdgcn_wave_barrier();
#endif

            // GLU: (a + b1a) * sigmoid(g + b1g)
#pragma unroll
            for (int nt = 0; nt < 2; ++nt) {
                v16b Ba = frag_B(Pw1a[w] + nt * 16 * 32, 32, 0);
                v16b Bg = frag_B(Pw1g[w] + nt * 16 * 32, 32, 0);
                int ncol = nt * 16 + nl;
                int ch = cb + ncol;
                float ba = b1[ch], bg = b1[CCH + ch];
#pragma unroll
                for (int mt = 0; mt < 4; ++mt) {
                    v8f za = {}, zg = {};
                    za = wmma_bf(Af[mt], Ba, za);
                    zg = wmma_bf(Af[mt], Bg, zg);
#pragma unroll
                    for (int r = 0; r < 8; ++r) {
                        int m = mt * 16 + r + hi8;
                        float a = za[r] + ba, g = zg[r] + bg;
                        Phg[w][m * 32 + ncol] = (__bf16)(a * sigf(g));
                    }
                }
            }
            __builtin_amdgcn_wave_barrier();

            // depthwise conv (9-tap, pad 4) + bias + BN + SiLU, rolling window
            int ch = cb + lane;
            float wdv[KW];
#pragma unroll
            for (int t = 0; t < KW; ++t) wdv[t] = Wd[ch * KW + t];
            float bdv = bd[ch];
            float sc = rsqrtf(v1[ch] + EPSC) * g1[ch];
            float mn = m1[ch], bb = be1[ch];
            float win[KW];
#pragma unroll
            for (int t = 0; t < KW; ++t) {
                int s = t - 4;
                win[t] = (s >= 0 && s < SEQ) ? (float)Phg[w][s * 32 + lane] : 0.f;
            }
            for (int l = 0; l < 64; ++l) {
                float o = 0.f;
                if (l < SEQ) {
#pragma unroll
                    for (int t = 0; t < KW; ++t) o += win[t] * wdv[t];
                    o += bdv;
                    o = (o - mn) * sc + bb;
                    o = siluf(o);
                }
                Psa[w][l * 32 + lane] = (__bf16)o;
#pragma unroll
                for (int t = 0; t < KW - 1; ++t) win[t] = win[t + 1];
                int s = l + 5;
                win[KW - 1] = (s < SEQ) ? (float)Phg[w][s * 32 + lane] : 0.f;
            }
            __builtin_amdgcn_wave_barrier();

            // y += S @ W2_chunk
            v16b Bw0 = frag_B(Pw2[w], 32, 0);
            v16b Bw1 = frag_B(Pw2[w] + 16 * 32, 32, 0);
#pragma unroll
            for (int mt = 0; mt < 4; ++mt) {
                v16b As = frag_A(Psa[w] + mt * 16 * 32, 32, 0);
                yfr[mt][0] = wmma_bf(As, Bw0, yfr[mt][0]);
                yfr[mt][1] = wmma_bf(As, Bw1, yfr[mt][1]);
            }
        }
    }

#pragma unroll
    for (int mt = 0; mt < 4; ++mt)
#pragma unroll
        for (int nt = 0; nt < 2; ++nt)
#pragma unroll
            for (int r = 0; r < 8; ++r)
                atomicAdd(&yacc[(mt * 16 + r + hi8) * 32 + nt * 16 + nl], yfr[mt][nt][r]);
    __syncthreads();

    // residual + LayerNorm -> X1
    if (tid < 64) {
        int l = tid;
        if (l < SEQ) {
            float vals[VOC]; float mean = 0.f;
#pragma unroll
            for (int v = 0; v < VOC; ++v) { vals[v] = Xf[l * 32 + v] + yacc[l * 32 + v]; mean += vals[v]; }
            mean *= (1.f / VOC);
            float var = 0.f;
#pragma unroll
            for (int v = 0; v < VOC; ++v) { float d = vals[v] - mean; var += d * d; }
            var *= (1.f / VOC);
            float inv = rsqrtf(var + EPSC);
            for (int v = 0; v < 32; ++v) {
                float o = (v < VOC) ? (vals[v] - mean) * inv * lng[v] + lnb[v] : 0.f;
                X1f[b * 2048 + l * 32 + v] = o;
                X1a[b * 2048 + l * 32 + v] = (__bf16)o;
            }
        } else {
            for (int v = 0; v < 32; ++v) {
                X1f[b * 2048 + l * 32 + v] = 0.f;
                X1a[b * 2048 + l * 32 + v] = (__bf16)0.f;
            }
        }
    }
}

// =======================================================================
// MHA + residual + LN -> flattened bf16 [B][1248]; 4 waves, head per wave.
// =======================================================================
__global__ __launch_bounds__(128) void k_mha(
    const float* __restrict__ X1f, const __bf16* __restrict__ X1a,
    const __bf16* __restrict__ Wqkvt, const __bf16* __restrict__ Wotp,
    const float* __restrict__ lng, const float* __restrict__ lnb,
    __bf16* __restrict__ Xflat) {
    __shared__ __align__(16) __bf16 Xs[64 * 32];
    __shared__ __align__(16) __bf16 Oa[64 * 192];
    __shared__ __align__(16) __bf16 Wot[32 * 192];
    __shared__ __align__(16) float  yln[64 * 32];
    __shared__ __align__(16) __bf16 Pq[4][64 * 32];
    __shared__ __align__(16) __bf16 Pk[4][64 * 32];
    __shared__ __align__(16) __bf16 Pvt[4][32 * 64];
    __shared__ __align__(16) float  Psc[4][16 * 64];
    __shared__ __align__(16) __bf16 Pat[4][64 * 64];
    __shared__ __align__(16) __bf16 Pbt[4][32 * 32];

    int b = blockIdx.x, tid = threadIdx.x, w = tid >> 5, lane = tid & 31;
    int nl = lane & 15, hi8 = (lane >> 4) << 3;
    const float scale = rsqrtf((float)VOC);

#if HAVE_TDM
    tdm_load_1d(Xs + w * 512, X1a + (size_t)b * 2048 + w * 512, 512);
    tdm_load_1d(Wot + w * 1536, Wotp + w * 1536, 1536);
    __builtin_amdgcn_s_wait_tensorcnt(0);
#else
    for (int i = tid; i < 2048; i += 128) Xs[i] = X1a[(size_t)b * 2048 + i];
    for (int i = tid; i < 32 * 192; i += 128) Wot[i] = Wotp[i];
#endif
    {
        v8b z8 = {};
        for (int i = tid; i < 64 * 192 / 8; i += 128) ((v8b*)Oa)[i] = z8;
    }
    __syncthreads();

    v16b Af[4];
#pragma unroll
    for (int mt = 0; mt < 4; ++mt) Af[mt] = frag_A(Xs + mt * 16 * 32, 32, 0);

    for (int hi = 0; hi < 3; ++hi) {
        int h = w + 4 * hi;                  // wave-uniform
        if (h < NHEAD) {
            for (int mtx = 0; mtx < 3; ++mtx) {
#if HAVE_TDM
                tdm_load_1d(Pbt[w], Wqkvt + (size_t)(mtx * NHEAD + h) * 1024, 1024);
                tdm_wait();
#else
                for (int i = lane; i < 1024; i += 32)
                    Pbt[w][i] = Wqkvt[(size_t)(mtx * NHEAD + h) * 1024 + i];
                __builtin_amdgcn_wave_barrier();
#endif
#pragma unroll
                for (int nt = 0; nt < 2; ++nt) {
                    v16b Bf = frag_B(Pbt[w] + nt * 16 * 32, 32, 0);
                    int d = nt * 16 + nl;
#pragma unroll
                    for (int mt = 0; mt < 4; ++mt) {
                        v8f c = {};
                        c = wmma_bf(Af[mt], Bf, c);
#pragma unroll
                        for (int r = 0; r < 8; ++r) {
                            int m = mt * 16 + r + hi8;
                            __bf16 val = (__bf16)c[r];
                            if (mtx == 0) Pq[w][m * 32 + d] = val;
                            else if (mtx == 1) Pk[w][m * 32 + d] = val;
                            else Pvt[w][d * 64 + m] = val;   // V^T [d][key]
                        }
                    }
                }
                __builtin_amdgcn_wave_barrier();
            }

            // scores = QK^T/sqrt(D), softmax over keys 0..58
            for (int mt = 0; mt < 4; ++mt) {
                v16b Aq = frag_A(Pq[w] + mt * 16 * 32, 32, 0);
#pragma unroll
                for (int nt = 0; nt < 4; ++nt) {
                    v16b Bk = frag_B(Pk[w] + nt * 16 * 32, 32, 0);
                    v8f c = {};
                    c = wmma_bf(Aq, Bk, c);
#pragma unroll
                    for (int r = 0; r < 8; ++r)
                        Psc[w][(r + hi8) * 64 + nt * 16 + nl] = c[r];
                }
                __builtin_amdgcn_wave_barrier();
                if (lane < 16) {
                    int row = lane;
                    float mx = -1e30f;
                    for (int k = 0; k < SEQ; ++k) {
                        float x = Psc[w][row * 64 + k] * scale;
                        if (x > mx) mx = x;
                    }
                    float s = 0.f;
                    for (int k = 0; k < SEQ; ++k) {
                        float e = __expf(Psc[w][row * 64 + k] * scale - mx);
                        s += e;
                        Psc[w][row * 64 + k] = e;
                    }
                    float inv = 1.f / s;
                    for (int k = 0; k < 64; ++k)
                        Pat[w][(mt * 16 + row) * 64 + k] =
                            (k < SEQ) ? (__bf16)(Psc[w][row * 64 + k] * inv) : (__bf16)0.f;
                }
                __builtin_amdgcn_wave_barrier();
            }

            // o = attn @ V
            for (int mt = 0; mt < 4; ++mt) {
#pragma unroll
                for (int dt = 0; dt < 2; ++dt) {
                    v8f c = {};
#pragma unroll
                    for (int ks = 0; ks < 2; ++ks) {
                        v16b Aa = frag_A(Pat[w] + mt * 16 * 64, 64, ks * 32);
                        v16b Bv = frag_B(Pvt[w] + dt * 16 * 64, 64, ks * 32);
                        c = wmma_bf(Aa, Bv, c);
                    }
                    int d = dt * 16 + nl;
                    if (d < VOC) {
#pragma unroll
                        for (int r = 0; r < 8; ++r) {
                            int m = mt * 16 + r + hi8;
                            Oa[m * 192 + h * VOC + d] = (__bf16)c[r];
                        }
                    }
                }
            }
        }
    }
    __syncthreads();

    // output projection
    for (int t2 = 0; t2 < 2; ++t2) {
        int task = w + 4 * t2;               // 0..7 -> (mt, nt)
        int mt = task >> 1, nt = task & 1;
        v8f c = {};
#pragma unroll
        for (int ks = 0; ks < 6; ++ks) {
            v16b Ao = frag_A(Oa + mt * 16 * 192, 192, ks * 32);
            v16b Bw = frag_B(Wot + nt * 16 * 192, 192, ks * 32);
            c = wmma_bf(Ao, Bw, c);
        }
#pragma unroll
        for (int r = 0; r < 8; ++r)
            yln[(mt * 16 + r + hi8) * 32 + nt * 16 + nl] = c[r];
    }
    __syncthreads();

    // residual + LN -> flattened bf16 row (padded to 1248)
    if (tid < SEQ) {
        int l = tid;
        float vals[VOC]; float mean = 0.f;
#pragma unroll
        for (int v = 0; v < VOC; ++v) {
            vals[v] = X1f[(size_t)b * 2048 + l * 32 + v] + yln[l * 32 + v];
            mean += vals[v];
        }
        mean *= (1.f / VOC);
        float var = 0.f;
#pragma unroll
        for (int v = 0; v < VOC; ++v) { float d = vals[v] - mean; var += d * d; }
        var *= (1.f / VOC);
        float inv = rsqrtf(var + EPSC);
#pragma unroll
        for (int v = 0; v < VOC; ++v)
            Xflat[(size_t)b * FLATP + l * VOC + v] =
                (__bf16)((vals[v] - mean) * inv * lng[v] + lnb[v]);
    }
    if (tid >= 64 && tid < 64 + (FLATP - FLATN))
        Xflat[(size_t)b * FLATP + FLATN + (tid - 64)] = (__bf16)0.f;
}

// =======================================================================
// head GEMM1: [1024 x 1248] @ Wf1t^T; B panel staged to LDS via TDM.
// grid (8, 10): 128 rows x 80 cols per WG.
// =======================================================================
__global__ __launch_bounds__(256) void k_head1(
    const __bf16* __restrict__ Xflat, const __bf16* __restrict__ Wf1t,
    const float* __restrict__ bf1,
    const float* __restrict__ g2, const float* __restrict__ be2,
    const float* __restrict__ m2, const float* __restrict__ v2,
    __bf16* __restrict__ h1) {
    __shared__ __align__(16) __bf16 Bs[80 * 32];
    int mb = blockIdx.x, nb = blockIdx.y;
    int tid = threadIdx.x, w = tid >> 5, lane = tid & 31;
    int nl = lane & 15, hi8 = (lane >> 4) << 3;

    const __bf16* Abase = Xflat + (size_t)(mb * 128 + w * 16) * FLATP;

    v8f acc[5] = {};
    for (int ks = 0; ks < FLATP / 32; ++ks) {
        // stage B panel [80 n][32 k]; wave w copies its 10 rows
#if HAVE_TDM
        tdm_load_2d(Bs + w * 10 * 32,
                    Wf1t + (size_t)(nb * 80 + w * 10) * FLATP + ks * 32,
                    32, 10, FLATP, 32, 10);
        __builtin_amdgcn_s_wait_tensorcnt(0);
#else
        for (int i = tid; i < 80 * 32; i += 256) {
            int n = i >> 5, k = i & 31;
            Bs[i] = Wf1t[(size_t)(nb * 80 + n) * FLATP + ks * 32 + k];
        }
#endif
        if (ks + 1 < FLATP / 32)
            __builtin_prefetch(Abase + lane * FLATP + (ks + 1) * 32, 0, 0);
        __syncthreads();
        v16b Aa = frag_A(Abase, FLATP, ks * 32);
#pragma unroll
        for (int nt = 0; nt < 5; ++nt) {
            v16b Bb = frag_B(Bs + nt * 16 * 32, 32, 0);
            acc[nt] = wmma_bf(Aa, Bb, acc[nt]);
        }
        __syncthreads();
    }
#pragma unroll
    for (int nt = 0; nt < 5; ++nt) {
        int n = nb * 80 + nt * 16 + nl;
        float bias = bf1[n];
        float sc = rsqrtf(v2[n] + EPSC) * g2[n];
        float mn = m2[n], bb = be2[n];
#pragma unroll
        for (int r = 0; r < 8; ++r) {
            int m = mb * 128 + w * 16 + r + hi8;
            float x = acc[nt][r] + bias;
            x = siluf(x);
            x = (x - mn) * sc + bb;
            h1[(size_t)m * NHID + n] = (__bf16)x;
        }
    }
}

// =======================================================================
// head GEMM2: relu([1024 x 800] @ Wf2t^T + bf2) -> h2 fp32 [1024 x 64]
// =======================================================================
__global__ __launch_bounds__(256) void k_head2(
    const __bf16* __restrict__ h1, const __bf16* __restrict__ Wf2t,
    const float* __restrict__ bf2, float* __restrict__ h2) {
    int mb = blockIdx.x;
    int tid = threadIdx.x, w = tid >> 5, lane = tid & 31;
    int nl = lane & 15, hi8 = (lane >> 4) << 3;

    const __bf16* Abase = h1 + (size_t)(mb * 128 + w * 16) * NHID;
    v8f acc[4] = {};
    for (int ks = 0; ks < NHID / 32; ++ks) {
        v16b Aa = frag_A(Abase, NHID, ks * 32);
#pragma unroll
        for (int nt = 0; nt < 4; ++nt) {
            v16b Bb = frag_B(Wf2t + nt * 16 * NHID, NHID, ks * 32);
            acc[nt] = wmma_bf(Aa, Bb, acc[nt]);
        }
    }
#pragma unroll
    for (int nt = 0; nt < 4; ++nt) {
        int n = nt * 16 + nl;
        float bias = bf2[n];
#pragma unroll
        for (int r = 0; r < 8; ++r) {
            int m = mb * 128 + w * 16 + r + hi8;
            float x = acc[nt][r] + bias;
            h2[(size_t)m * 64 + n] = (x > 0.f) ? x : 0.f;
        }
    }
}

// =======================================================================
// final: out = h2 @ Wf3 + bf3
// =======================================================================
__global__ __launch_bounds__(256) void k_final(
    const float* __restrict__ h2, const float* __restrict__ Wf3,
    const float* __restrict__ bf3, float* __restrict__ out) {
    int b = blockIdx.x * 256 + threadIdx.x;
    if (b < BSZ) {
        float a0 = bf3[0], a1 = bf3[1];
#pragma unroll 8
        for (int i = 0; i < 64; ++i) {
            float x = h2[b * 64 + i];
            a0 += x * Wf3[i * 2 + 0];
            a1 += x * Wf3[i * 2 + 1];
        }
        out[b * 2 + 0] = a0;
        out[b * 2 + 1] = a1;
    }
}

// =======================================================================
extern "C" void kernel_launch(void* const* d_in, const int* in_sizes, int n_in,
                              void* d_out, int out_size, void* d_ws, size_t ws_size,
                              hipStream_t stream) {
    (void)in_sizes; (void)n_in; (void)out_size; (void)ws_size;
    const float* pep  = (const float*)d_in[0];
    const float* mhc  = (const float*)d_in[1];
    const float* W1   = (const float*)d_in[2];
    const float* b1   = (const float*)d_in[3];
    const float* Wd   = (const float*)d_in[4];
    const float* bd   = (const float*)d_in[5];
    const float* bn1g = (const float*)d_in[6];
    const float* bn1b = (const float*)d_in[7];
    const float* bn1m = (const float*)d_in[8];
    const float* bn1v = (const float*)d_in[9];
    const float* W2   = (const float*)d_in[10];
    const float* b2   = (const float*)d_in[11];
    const float* lng  = (const float*)d_in[12];
    const float* lnb  = (const float*)d_in[13];
    const float* Wq   = (const float*)d_in[14];
    const float* Wk   = (const float*)d_in[15];
    const float* Wv   = (const float*)d_in[16];
    const float* Wo   = (const float*)d_in[17];
    const float* Wf1  = (const float*)d_in[18];
    const float* bf1  = (const float*)d_in[19];
    const float* bn2g = (const float*)d_in[20];
    const float* bn2b = (const float*)d_in[21];
    const float* bn2m = (const float*)d_in[22];
    const float* bn2v = (const float*)d_in[23];
    const float* Wf2  = (const float*)d_in[24];
    const float* bf2  = (const float*)d_in[25];
    const float* Wf3  = (const float*)d_in[26];
    const float* bf3  = (const float*)d_in[27];

    char* p = (char*)d_ws;
    float*  X1f   = (float*)p;   p += (size_t)BSZ * 2048 * 4;
    __bf16* X1a   = (__bf16*)p;  p += (size_t)BSZ * 2048 * 2;
    __bf16* Xflat = (__bf16*)p;  p += (size_t)BSZ * FLATP * 2;
    __bf16* Wf1t  = (__bf16*)p;  p += (size_t)NHID * FLATP * 2;
    __bf16* Wf2t  = (__bf16*)p;  p += (size_t)64 * NHID * 2;
    __bf16* h1    = (__bf16*)p;  p += (size_t)BSZ * NHID * 2;
    float*  h2    = (float*)p;   p += (size_t)BSZ * 64 * 4;
    __bf16* W1p   = (__bf16*)p;  p += (size_t)6400 * 32 * 2;
    __bf16* W2t   = (__bf16*)p;  p += (size_t)32 * CCH * 2;
    __bf16* Wqkvt = (__bf16*)p;  p += (size_t)3 * NHEAD * 1024 * 2;
    __bf16* Wotp  = (__bf16*)p;  p += (size_t)32 * 192 * 2;

    const int prepN = (PA6 + 255) / 256;
    k_prep<<<prepN, 256, 0, stream>>>(W1, W2, Wq, Wk, Wv, Wo, Wf1, Wf2,
                                      W1p, W2t, Wqkvt, Wotp, Wf1t, Wf2t);
    k_conv<<<BSZ, 256, 0, stream>>>(pep, mhc, W1p, b1, Wd, bd, bn1g, bn1b, bn1m, bn1v,
                                    W2t, b2, lng, lnb, X1f, X1a);
    k_mha<<<BSZ, 128, 0, stream>>>(X1f, X1a, Wqkvt, Wotp, lng, lnb, Xflat);
    k_head1<<<dim3(8, 10), 256, 0, stream>>>(Xflat, Wf1t, bf1, bn2g, bn2b, bn2m, bn2v, h1);
    k_head2<<<8, 256, 0, stream>>>(h1, Wf2t, bf2, h2);
    k_final<<<(BSZ + 255) / 256, 256, 0, stream>>>(h2, Wf3, bf3, (float*)d_out);
}
